// Attention_15822659518687
// MI455X (gfx1250) — matrix operand order repaired and checked by reading, not compiled, not run on hardware
//
#include <hip/hip_runtime.h>
#include <hip/hip_bf16.h>
#include <cstdint>

typedef __attribute__((ext_vector_type(16))) _Float16 v16h;
typedef __attribute__((ext_vector_type(8)))  _Float16 v8h;
typedef __attribute__((ext_vector_type(8)))  float    v8f;
typedef __attribute__((ext_vector_type(4)))  float    v4f;
typedef __attribute__((ext_vector_type(4)))  unsigned int u32x4;
typedef __attribute__((ext_vector_type(8)))  int      i32x8;
typedef __attribute__((ext_vector_type(4)))  int      i32x4;

union V16 { v16h v; v8h h[2]; };

#define SEQ   2048
#define DIM   4096
#define NH    32
#define NKV   8
#define HD    128
#define KVDIM 1024

// ---------------------------------------------------------------------------
// f32 -> f16 bulk convert (8 elements/thread). One-time pass so the f16
// weights (80 MB total) live in the 192 MB L2 across all GEMM row-blocks.
// ---------------------------------------------------------------------------
__global__ __launch_bounds__(256) void cvt_f16(const float* __restrict__ src,
                                               _Float16* __restrict__ dst, int n8)
{
  const int i = blockIdx.x * blockDim.x + threadIdx.x;
  if (i >= n8) return;
  const v4f a = ((const v4f*)src)[2 * i];
  const v4f b = ((const v4f*)src)[2 * i + 1];
  v8h o;
#pragma unroll
  for (int j = 0; j < 4; j++) { o[j] = (_Float16)a[j]; o[4 + j] = (_Float16)b[j]; }
  ((v8h*)dst)[i] = o;
}

// ---------------------------------------------------------------------------
// Tensor Data Mover: 2-D tile (tile_d0 halves x tile_d1 rows) from global
// (row stride = stride_elems halves) into LDS at lds_byte_off, with LDS
// padding 16 data DWORDs + 4 pad DWORDs -> 40-half (80 B) padded rows.
// Descriptor packing per CDNA5 ISA ch.8 (D# groups 0/1; groups 2/3 unused).
// Toolchain uses the 6-arg builtin form (extra unused i32x8 group + cpol).
// ---------------------------------------------------------------------------
__device__ __forceinline__ uint32_t lds_off(const void* p)
{
  return (uint32_t)(uintptr_t)p;   // LDS aperture: addr[31:0] = LDS byte offset
}

__device__ __forceinline__ void tdm_load_2d(const _Float16* gptr, uint32_t ldsoff,
                                            uint32_t tile_d0, uint32_t tile_d1,
                                            uint32_t stride_elems)
{
  const uint64_t ga = (uint64_t)(uintptr_t)gptr;
  u32x4 g0;
  g0[0] = 1u;                                            // count=1, user mode
  g0[1] = ldsoff;                                        // lds_addr
  g0[2] = (uint32_t)ga;                                  // global_addr[31:0]
  g0[3] = (uint32_t)((ga >> 32) & 0x1ffffffu) | (2u << 30); // addr[56:32] | type=2

  const uint32_t td0 = stride_elems;                     // tensor_dim0
  const uint32_t td1 = 1u << 20;                         // tensor_dim1 (ample)
  i32x8 g1;
  // data_size=1 (2B) | pad_enable | pad_interval=3 (16 dw) | pad_amount=3 (4 dw)
  g1[0] = (int)((1u << 16) | (1u << 20) | (3u << 22) | (3u << 25));
  g1[1] = (int)((td0 & 0xffffu) << 16);                  // tensor_dim0[15:0]
  g1[2] = (int)((td0 >> 16) | ((td1 & 0xffffu) << 16));  // td0[31:16] | td1[15:0]
  g1[3] = (int)((td1 >> 16) | (tile_d0 << 16));          // td1[31:16] | tile_dim0
  g1[4] = (int)(tile_d1);                                // tile_dim1 | tile_dim2=0
  g1[5] = (int)stride_elems;                             // tensor_dim0_stride[31:0]
  g1[6] = 0;                                             // stride[47:32] | dim1_stride lo
  g1[7] = 0;
  const i32x4 z4 = {0, 0, 0, 0};
  const i32x8 z8 = {0, 0, 0, 0, 0, 0, 0, 0};
  __builtin_amdgcn_tensor_load_to_lds(g0, g1, z4, z4, z8, 0);
}

// ---------------------------------------------------------------------------
// C[M][N] = A[M][K] @ W[N][K]^T, both operands f16 row-major. LDS staging via
// TDM, double-buffered: wave 0 issues tensor loads for step t+1, waits
// s_wait_tensorcnt(2) (in-order TDM completion => step t landed), barrier
// publishes. 8 waves (4Mx2N), wave tile 32x64, 8 WMMAs per k-step of 32.
// ---------------------------------------------------------------------------
template<bool OUT_HALF>
__global__ __launch_bounds__(256) void gemm_xwt(const _Float16* __restrict__ A,
                                                const _Float16* __restrict__ W,
                                                void* __restrict__ Cp,
                                                int M, int N, int K)
{
  __shared__ _Float16 As[2][128 * 40];   // 40-half padded rows (TDM pad)
  __shared__ _Float16 Ws[2][128 * 40];

  const int tid  = threadIdx.x;
  const int lane = tid & 31;
  const int wave = tid >> 5;
  const int g    = lane >> 4;
  const int ln   = lane & 15;
  const int m0 = blockIdx.y * 128;
  const int n0 = blockIdx.x * 128;
  const int waveM = wave >> 1;
  const int waveN = wave & 1;

  v8f acc[2][4];
#pragma unroll
  for (int i = 0; i < 2; i++)
#pragma unroll
    for (int j = 0; j < 4; j++) acc[i][j] = (v8f)0.f;

  const int nsteps = K >> 5;
  if (wave == 0) {
    tdm_load_2d(A + (size_t)m0 * K, lds_off(As[0]), 32, 128, (uint32_t)K);
    tdm_load_2d(W + (size_t)n0 * K, lds_off(Ws[0]), 32, 128, (uint32_t)K);
  }

  for (int t = 0; t < nsteps; ++t) {
    __syncthreads();                    // everyone done with buf[(t+1)&1]
    if (wave == 0) {
      if (t + 1 < nsteps) {
        const int k1 = (t + 1) << 5;
        tdm_load_2d(A + (size_t)m0 * K + k1, lds_off(As[(t + 1) & 1]), 32, 128, (uint32_t)K);
        tdm_load_2d(W + (size_t)n0 * K + k1, lds_off(Ws[(t + 1) & 1]), 32, 128, (uint32_t)K);
        __builtin_amdgcn_s_wait_tensorcnt(2);   // step t's two loads complete
      } else {
        __builtin_amdgcn_s_wait_tensorcnt(0);
      }
    }
    __syncthreads();                    // tile t visible to all waves

    const _Float16* as = As[t & 1];
    const _Float16* ws = Ws[t & 1];
    V16 a[2], b[4];
#pragma unroll
    for (int fm = 0; fm < 2; fm++) {
      const _Float16* ap = &as[(waveM * 32 + fm * 16 + ln) * 40];
      a[fm].h[0] = *(const v8h*)(ap + g * 8);
      a[fm].h[1] = *(const v8h*)(ap + 16 + g * 8);
    }
#pragma unroll
    for (int fn = 0; fn < 4; fn++) {
      const _Float16* bp = &ws[(waveN * 64 + fn * 16 + ln) * 40 + g * 8];
      b[fn].h[0] = ((const v8h*)bp)[0];
      b[fn].h[1] = ((const v8h*)bp)[2];
    }
#pragma unroll
    for (int fm = 0; fm < 2; fm++)
#pragma unroll
      for (int fn = 0; fn < 4; fn++)
        acc[fm][fn] = __builtin_amdgcn_wmma_f32_16x16x32_f16(
            false, a[fm].v, false, b[fn].v, (short)0, acc[fm][fn], false, false);
  }

#pragma unroll
  for (int fm = 0; fm < 2; fm++) {
#pragma unroll
    for (int fn = 0; fn < 4; fn++) {
      const int col = n0 + waveN * 64 + fn * 16 + ln;
#pragma unroll
      for (int r = 0; r < 8; r++) {
        const int row = m0 + waveM * 32 + fm * 16 + r + 8 * g;
        const float val = acc[fm][fn][r];
        if constexpr (OUT_HALF)
          ((_Float16*)Cp)[(size_t)row * N + col] = (_Float16)val;
        else
          ((float*)Cp)[(size_t)row * N + col] = val;
      }
    }
  }
}

// ---------------------------------------------------------------------------
// RoPE (interleaved pairs), in-place on f16 activations.
// ---------------------------------------------------------------------------
__global__ void rope_k(_Float16* q, int stride, int nheads)
{
  const int idx   = blockIdx.x * blockDim.x + threadIdx.x;
  const int total = SEQ * nheads * 64;
  if (idx >= total) return;
  const int i = idx & 63;
  const int h = (idx >> 6) % nheads;
  const int p = idx / (nheads * 64);
  const float freq = __expf(-((2.0f * i) / 128.0f) * __logf(10000.0f));
  float s, c;
  __sincosf((float)p * freq, &s, &c);
  _Float16* base = q + (size_t)p * stride + h * HD + 2 * i;
  const float xr = (float)base[0], xi = (float)base[1];
  base[0] = (_Float16)(xr * c - xi * s);
  base[1] = (_Float16)(xr * s + xi * c);
}

// ---------------------------------------------------------------------------
// Flash attention, causal, GQA 4:1. Block = (head, 128 q-rows), 8 waves,
// each wave owns 16 q-rows. KV walked in 32-wide steps (block-uniform bound
// so __syncthreads around shared K/V staging is safe). Per-wave softmax
// round-trips through private LDS with s_wait_dscnt ordering.
// ---------------------------------------------------------------------------
__global__ __launch_bounds__(256) void flash_attn(const _Float16* __restrict__ q,
                                                  const _Float16* __restrict__ k,
                                                  const _Float16* __restrict__ v,
                                                  _Float16* __restrict__ o)
{
  __shared__ _Float16 Kt[32 * 136];
  __shared__ _Float16 Vt[128 * 40];
  __shared__ float    SL[8][16 * 36];
  __shared__ _Float16 PL[8][16 * 40];
  __shared__ float    AL[8][16];
  __shared__ float    LL[8][16];

  const int tid  = threadIdx.x;
  const int lane = tid & 31;
  const int wave = tid >> 5;
  const int g    = lane >> 4;
  const int ln   = lane & 15;
  const int head = blockIdx.y;
  const int kvh  = head >> 2;
  const int qblock = blockIdx.x;
  const int qbase  = qblock * 128 + wave * 16;
  const int kvend  = qblock * 128 + 128;

  V16 qa[4];
  {
    const _Float16* qp = q + (size_t)(qbase + ln) * DIM + head * HD;
#pragma unroll
    for (int d0 = 0; d0 < 4; d0++) {
      qa[d0].h[0] = *(const v8h*)(qp + d0 * 32 + g * 8);
      qa[d0].h[1] = *(const v8h*)(qp + d0 * 32 + 16 + g * 8);
    }
  }

  v8f oacc[8];
#pragma unroll
  for (int i = 0; i < 8; i++) oacc[i] = (v8f)0.f;
  float m_run = -3.0e38f, l_run = 0.f;
  const float scale = 0.08838834764831845f;  // 1/sqrt(128)

  for (int kv0 = 0; kv0 < kvend; kv0 += 32) {
    __syncthreads();
    {
      const int row = tid >> 3, c16 = tid & 7;
      const _Float16* src = k + (size_t)(kv0 + row) * KVDIM + kvh * HD + c16 * 16;
      _Float16* dst = &Kt[row * 136 + c16 * 16];
      ((v8h*)dst)[0] = ((const v8h*)src)[0];
      ((v8h*)dst)[1] = ((const v8h*)src)[1];
    }
    {
      const int kr = tid & 31, dc = (tid >> 5) * 16;
      const _Float16* src = v + (size_t)(kv0 + kr) * KVDIM + kvh * HD + dc;
      const v8h s0 = ((const v8h*)src)[0];
      const v8h s1 = ((const v8h*)src)[1];
#pragma unroll
      for (int i = 0; i < 8; i++) Vt[(dc + i) * 40 + kr] = s0[i];
#pragma unroll
      for (int i = 0; i < 8; i++) Vt[(dc + 8 + i) * 40 + kr] = s1[i];
    }
    __syncthreads();

#pragma unroll
    for (int t = 0; t < 2; t++) {
      v8f s = (v8f)0.f;
#pragma unroll
      for (int d0 = 0; d0 < 4; d0++) {
        V16 b;
        const _Float16* bp = &Kt[(t * 16 + ln) * 136 + d0 * 32 + g * 8];
        b.h[0] = ((const v8h*)bp)[0];
        b.h[1] = ((const v8h*)bp)[2];
        s = __builtin_amdgcn_wmma_f32_16x16x32_f16(
            false, qa[d0].v, false, b.v, (short)0, s, false, false);
      }
#pragma unroll
      for (int r = 0; r < 8; r++)
        SL[wave][(r + 8 * g) * 36 + t * 16 + ln] = s[r];
    }
    asm volatile("s_wait_dscnt 0" ::: "memory");

    if (lane < 16) {
      const int row  = lane;
      const int qpos = qbase + row;
      const float mold = m_run;
      float tm = mold;
      float sv[32];
#pragma unroll
      for (int c = 0; c < 32; c++) {
        const int kvpos = kv0 + c;
        float x = SL[wave][row * 36 + c] * scale;
        x = (kvpos <= qpos) ? x : -3.0e38f;
        sv[c] = x;
        tm = fmaxf(tm, x);
      }
      const float alpha = __expf(mold - tm);
      float sum = 0.f;
#pragma unroll
      for (int c = 0; c < 32; c++) {
        const float p = __expf(sv[c] - tm);
        sum += p;
        PL[wave][row * 40 + c] = (_Float16)p;
      }
      l_run = l_run * alpha + sum;
      m_run = tm;
      AL[wave][row] = alpha;
    }
    asm volatile("s_wait_dscnt 0" ::: "memory");

    {
      const v4f a0 = *(const v4f*)&AL[wave][8 * g];
      const v4f a1 = *(const v4f*)&AL[wave][8 * g + 4];
#pragma unroll
      for (int nf = 0; nf < 8; nf++) {
#pragma unroll
        for (int r = 0; r < 4; r++) oacc[nf][r] *= a0[r];
#pragma unroll
        for (int r = 0; r < 4; r++) oacc[nf][4 + r] *= a1[r];
      }
    }

    V16 pa;
    {
      const _Float16* pp = &PL[wave][ln * 40];
      pa.h[0] = *(const v8h*)(pp + g * 8);
      pa.h[1] = *(const v8h*)(pp + 16 + g * 8);
    }
#pragma unroll
    for (int nf = 0; nf < 8; nf++) {
      V16 b;
      const _Float16* bp = &Vt[(nf * 16 + ln) * 40 + g * 8];
      b.h[0] = ((const v8h*)bp)[0];
      b.h[1] = ((const v8h*)bp)[2];
      oacc[nf] = __builtin_amdgcn_wmma_f32_16x16x32_f16(
          false, pa.v, false, b.v, (short)0, oacc[nf], false, false);
    }
  }

  if (lane < 16) LL[wave][lane] = l_run;
  asm volatile("s_wait_dscnt 0" ::: "memory");
  {
    const v4f l0 = *(const v4f*)&LL[wave][8 * g];
    const v4f l1 = *(const v4f*)&LL[wave][8 * g + 4];
    float inv[8];
#pragma unroll
    for (int r = 0; r < 4; r++) { inv[r] = 1.0f / l0[r]; inv[4 + r] = 1.0f / l1[r]; }
#pragma unroll
    for (int nf = 0; nf < 8; nf++) {
      const int col = head * HD + nf * 16 + ln;
#pragma unroll
      for (int r = 0; r < 8; r++) {
        const int row = qbase + r + 8 * g;
        o[(size_t)row * DIM + col] = (_Float16)(oacc[nf][r] * inv[r]);
      }
    }
  }
}

// ---------------------------------------------------------------------------
extern "C" void kernel_launch(void* const* d_in, const int* in_sizes, int n_in,
                              void* d_out, int out_size, void* d_ws, size_t ws_size,
                              hipStream_t stream)
{
  (void)in_sizes; (void)n_in; (void)out_size; (void)ws_size;
  const float* x  = (const float*)d_in[0];
  const float* wq = (const float*)d_in[2];
  const float* wk = (const float*)d_in[3];
  const float* wv = (const float*)d_in[4];
  const float* wo = (const float*)d_in[5];
  float* out = (float*)d_out;

  char* ws = (char*)d_ws;
  const size_t MB = 1024 * 1024;
  _Float16* q16  = (_Float16*)(ws);              // 16MB
  _Float16* k16  = (_Float16*)(ws + 16 * MB);    //  4MB
  _Float16* v16  = (_Float16*)(ws + 20 * MB);    //  4MB
  _Float16* a16  = (_Float16*)(ws + 24 * MB);    // 16MB
  _Float16* x16  = (_Float16*)(ws + 40 * MB);    // 16MB
  _Float16* wq16 = (_Float16*)(ws + 56 * MB);    // 32MB
  _Float16* wk16 = (_Float16*)(ws + 88 * MB);    //  8MB
  _Float16* wv16 = (_Float16*)(ws + 96 * MB);    //  8MB
  _Float16* wo16 = (_Float16*)(ws + 104 * MB);   // 32MB

  const dim3 blk(256);
  // one-time f32->f16 conversions (f16 weights total 80MB -> resident in L2)
  cvt_f16<<<(SEQ * DIM / 8 + 255) / 256, blk, 0, stream>>>(x,  x16,  SEQ * DIM / 8);
  cvt_f16<<<(DIM * DIM / 8 + 255) / 256, blk, 0, stream>>>(wq, wq16, DIM * DIM / 8);
  cvt_f16<<<(KVDIM * DIM / 8 + 255) / 256, blk, 0, stream>>>(wk, wk16, KVDIM * DIM / 8);
  cvt_f16<<<(KVDIM * DIM / 8 + 255) / 256, blk, 0, stream>>>(wv, wv16, KVDIM * DIM / 8);
  cvt_f16<<<(DIM * DIM / 8 + 255) / 256, blk, 0, stream>>>(wo, wo16, DIM * DIM / 8);
  // QKV projections (TDM-staged f16 GEMMs, f16 out)
  gemm_xwt<true><<<dim3(DIM / 128,   SEQ / 128), blk, 0, stream>>>(x16, wq16, q16, SEQ, DIM,   DIM);
  gemm_xwt<true><<<dim3(KVDIM / 128, SEQ / 128), blk, 0, stream>>>(x16, wk16, k16, SEQ, KVDIM, DIM);
  gemm_xwt<true><<<dim3(KVDIM / 128, SEQ / 128), blk, 0, stream>>>(x16, wv16, v16, SEQ, KVDIM, DIM);
  // RoPE
  rope_k<<<(SEQ * NH  * 64 + 255) / 256, blk, 0, stream>>>(q16, DIM,   NH);
  rope_k<<<(SEQ * NKV * 64 + 255) / 256, blk, 0, stream>>>(k16, KVDIM, NKV);
  // Causal GQA flash attention
  flash_attn<<<dim3(SEQ / 128, NH), blk, 0, stream>>>(q16, k16, v16, a16);
  // Output projection (f32 out)
  gemm_xwt<false><<<dim3(DIM / 128, SEQ / 128), blk, 0, stream>>>(a16, wo16, out, SEQ, DIM, DIM);
}
